// CrossAttention_19963007992389
// MI455X (gfx1250) — matrix operand order repaired and checked by reading, not compiled, not run on hardware
//
#include <hip/hip_runtime.h>
#include <hip/hip_bf16.h>

// ---------------- problem constants (match reference) ----------------
constexpr int DIM   = 1024;
constexpr int CTX   = 1024;
constexpr int HEADS = 12;
constexpr int DHEAD = 64;
constexpr int INNER = HEADS * DHEAD;   // 768
constexpr int B_    = 4;
constexpr int F_    = 2048;
constexpr int J_    = 1024;

typedef __attribute__((ext_vector_type(16))) _Float16   v16h;
typedef __attribute__((ext_vector_type(8)))  float      v8f;
typedef __attribute__((ext_vector_type(4)))  unsigned   uint4v;

// exact pointee type from hipcc diagnostic: vector_size(16) int
typedef int v4i_t __attribute__((vector_size(16)));
typedef __attribute__((address_space(1))) v4i_t g_v4i;   // global int4
typedef __attribute__((address_space(3))) v4i_t l_v4i;   // LDS int4
typedef __attribute__((address_space(3))) void  lvoid;   // LDS void

// ---------------- async global->LDS staging (gfx1250 LDS-DMA path) ----------------
#if __has_builtin(__builtin_amdgcn_global_load_async_to_lds_b128)
  #define ASYNC_MODE 2
#elif __has_builtin(__builtin_amdgcn_load_to_lds)
  #define ASYNC_MODE 1
#else
  #define ASYNC_MODE 0
#endif

__device__ __forceinline__ void copy16_to_lds(const _Float16* g, _Float16* l) {
#if ASYNC_MODE == 2
    __builtin_amdgcn_global_load_async_to_lds_b128(
        (g_v4i*)(unsigned long long)g,
        (l_v4i*)(unsigned)(unsigned long long)l, 0, 0);
#elif ASYNC_MODE == 1
    __builtin_amdgcn_load_to_lds((void*)g,
        (lvoid*)(unsigned)(unsigned long long)l, 16, 0, 0);
#else
    *(uint4v*)l = *(const uint4v*)g;    // sync fallback (register bounce)
#endif
}

template<int N>
__device__ __forceinline__ void wait_async() {
#if ASYNC_MODE
  #if __has_builtin(__builtin_amdgcn_s_wait_asynccnt)
    __builtin_amdgcn_s_wait_asynccnt(N);
  #else
    asm volatile("s_wait_asynccnt %0" :: "n"(N) : "memory");
  #endif
#endif
}

// ---------------- WMMA helper (CDNA5 16x16x32 f16 -> f32) ----------------
__device__ __forceinline__ v8f wmma16(v16h a, v16h b, v8f c) {
    // (neg_a, A, neg_b, B, c_mod, C, reuse_a, reuse_b)
    return __builtin_amdgcn_wmma_f32_16x16x32_f16(false, a, false, b, (short)0, c,
                                                  false, false);
}

// A fragment, 16x32 (MxK), f16, row-major source with leading dim `ld`.
// Lane L: row m = L&15, half = L>>4.  VGPR v<4 -> K = 2v+8*half (+0,1)
//                                     VGPR v>=4 -> K = 16+2(v-4)+8*half
__device__ __forceinline__ v16h load_fragA(const _Float16* base, int ld, int row, int k0) {
    const int lane = threadIdx.x & 31;
    const int m = lane & 15, half = lane >> 4;
    const _Float16* p = base + (size_t)(row + m) * ld + k0 + 8 * half;
    union { v16h h; uint4v q[2]; } f;
    f.q[0] = *(const uint4v*)(p);
    f.q[1] = *(const uint4v*)(p + 16);
    return f.h;
}

// B fragment, 32x16 (KxN), f16, loaded from TRANSPOSED storage B^T[N][K], ld = K-stride.
// Lane L: col n = L&15, half = L>>4. Elements 0-7 K=8*half+0..7, elements 8-15 K=16+8*half+0..7.
__device__ __forceinline__ v16h load_fragB(const _Float16* baseT, int ld, int col, int k0) {
    const int lane = threadIdx.x & 31;
    const int n = lane & 15, half = lane >> 4;
    const _Float16* p = baseT + (size_t)(col + n) * ld + k0 + 8 * half;
    union { v16h h; uint4v q[2]; } f;
    f.q[0] = *(const uint4v*)(p);
    f.q[1] = *(const uint4v*)(p + 16);
    return f.h;
}

// 16-lane max reduction via DPP ROW_XMASK (pure VALU, no LDS traffic).
#define DPP_XMASK_MAX(v, ctrl)                                                     \
    v = fmaxf(v, __int_as_float(__builtin_amdgcn_update_dpp(                       \
                0, __float_as_int(v), (ctrl), 0xf, 0xf, true)))
__device__ __forceinline__ float redmax16(float v) {
    DPP_XMASK_MAX(v, 0x161);   // row_xmask xor 1
    DPP_XMASK_MAX(v, 0x162);   // row_xmask xor 2
    DPP_XMASK_MAX(v, 0x164);   // row_xmask xor 4
    DPP_XMASK_MAX(v, 0x168);   // row_xmask xor 8
    return v;
}

__device__ __forceinline__ v16h ones_frag() {
    union { v16h h; _Float16 e[16]; } o;
#pragma unroll
    for (int i = 0; i < 16; ++i) o.e[i] = (_Float16)1.0f;
    return o.h;
}

// ---------------- weight convert + transpose: WT[n*K+k] = (f16)W[k*N+n] ----------------
__global__ __launch_bounds__(256) void wT_kernel(const float* __restrict__ W,
                                                 _Float16* __restrict__ WT,
                                                 int K, int N) {
    int i = blockIdx.x * 256 + threadIdx.x;          // over N*K
    if (i >= N * K) return;
    int k = i % K, n = i / K;
    WT[(size_t)n * K + k] = (_Float16)W[(size_t)k * N + n];
}

// ---------------- row layernorm: f32 in -> f16 out, optional post-scale ----------------
__global__ __launch_bounds__(256) void ln_kernel(const float* __restrict__ in,
                                                 _Float16* __restrict__ out,
                                                 const float* __restrict__ gw,
                                                 const float* __restrict__ bw,
                                                 int cols, float scale) {
    __shared__ float r1[256], r2[256];
    const size_t row = blockIdx.x;
    const float* rp = in + row * (size_t)cols;
    float s = 0.f, s2 = 0.f;
    for (int c = threadIdx.x; c < cols; c += 256) { float v = rp[c]; s += v; s2 += v * v; }
    r1[threadIdx.x] = s; r2[threadIdx.x] = s2;
    __syncthreads();
    for (int off = 128; off > 0; off >>= 1) {
        if ((int)threadIdx.x < off) {
            r1[threadIdx.x] += r1[threadIdx.x + off];
            r2[threadIdx.x] += r2[threadIdx.x + off];
        }
        __syncthreads();
    }
    float mu  = r1[0] / cols;
    float var = r2[0] / cols - mu * mu;
    float inv = rsqrtf(var + 1e-5f);
    _Float16* op = out + row * (size_t)cols;
    for (int c = threadIdx.x; c < cols; c += 256)
        op[c] = (_Float16)(((rp[c] - mu) * inv * gw[c] + bw[c]) * scale);
}

// ---------------- v transpose: [B*J, INNER] f32 -> [B, INNER, J] f16 ----------------
__global__ __launch_bounds__(256) void vT_kernel(const float* __restrict__ vp,
                                                 _Float16* __restrict__ vT) {
    size_t i = (size_t)blockIdx.x * 256 + threadIdx.x;   // over B*J*INNER
    if (i >= (size_t)B_ * J_ * INNER) return;
    int inner = (int)(i % INNER);
    size_t rj = i / INNER;
    int j = (int)(rj % J_);
    int b = (int)(rj / J_);
    vT[((size_t)b * INNER + inner) * J_ + j] = (_Float16)vp[i];
}

// ---------------- WMMA GEMM with double-buffered async LDS staging --------------------
// C[M,N](f32) = A[M,K](f16) @ BT[N,K]^T (+bias).
// block = 256 thr = 8 waves; wave (wr,wc) owns 32x64 (2x4 frags); block tile 64 x 256.
// Per k-step: stage A slab (64x32, 4KB) + BT slab (256x32, 16KB) into LDS via
// ASYNCcnt-tracked LDS-DMA (5 b128 per thread), double buffered.
template<bool BIAS>
__global__ __launch_bounds__(256) void gemm_kernel(const _Float16* __restrict__ A,
                                                   const _Float16* __restrict__ BT,
                                                   float* __restrict__ C,
                                                   const float* __restrict__ bias,
                                                   int M, int N, int K) {
    __shared__ __align__(16) _Float16 sa[2][64 * 32];    // 2 x 4KB
    __shared__ __align__(16) _Float16 sb[2][256 * 32];   // 2 x 16KB
    const int t  = threadIdx.x;
    const int w  = t >> 5;
    const int wr = w >> 2, wc = w & 3;
    const int row0 = blockIdx.y * 64;
    const int col0 = blockIdx.x * 256;

    auto stage = [&](int buf, int k0) {
        {   // A slab: 256 segments of 16B, 1 per thread
            int row = t >> 2, part = t & 3;
            copy16_to_lds(A + (size_t)(row0 + row) * K + k0 + part * 8,
                          &sa[buf][row * 32 + part * 8]);
        }
#pragma unroll
        for (int i = 0; i < 4; ++i) {   // BT slab: 1024 segments, 4 per thread
            int seg = t + i * 256;
            int row = seg >> 2, part = seg & 3;
            copy16_to_lds(BT + (size_t)(col0 + row) * K + k0 + part * 8,
                          &sb[buf][row * 32 + part * 8]);
        }
    };

    stage(0, 0);
    v8f acc0[4] = {{}, {}, {}, {}};
    v8f acc1[4] = {{}, {}, {}, {}};
    int buf = 0;
    for (int k0 = 0; k0 < K; k0 += 32) {
        if (k0 + 32 < K) {
            stage(buf ^ 1, k0 + 32);     // prefetch next slab while waiting on current
            wait_async<5>();             // async ops complete in order: oldest 5 done
        } else {
            wait_async<0>();
        }
        __syncthreads();                 // all waves' staged data visible

        v16h a0 = load_fragA(&sa[buf][0], 32, wr * 32,      0);
        v16h a1 = load_fragA(&sa[buf][0], 32, wr * 32 + 16, 0);
#pragma unroll
        for (int j = 0; j < 4; ++j) {
            v16h b = load_fragB(&sb[buf][0], 32, wc * 64 + 16 * j, 0);
            acc0[j] = wmma16(a0, b, acc0[j]);
            acc1[j] = wmma16(a1, b, acc1[j]);
        }
        __syncthreads();                 // reads done before buffer is re-staged
        buf ^= 1;
    }

    const int lane = threadIdx.x & 31;
    const int n = lane & 15, half = lane >> 4;
#pragma unroll
    for (int j = 0; j < 4; ++j) {
        float bv = BIAS ? bias[col0 + wc * 64 + 16 * j + n] : 0.f;
#pragma unroll
        for (int r = 0; r < 8; ++r) {
            int rr = r + 8 * half;
            C[(size_t)(row0 + wr * 32 + rr)      * N + col0 + wc * 64 + 16 * j + n] = acc0[j][r] + bv;
            C[(size_t)(row0 + wr * 32 + 16 + rr) * N + col0 + wc * 64 + 16 * j + n] = acc1[j][r] + bv;
        }
    }
}

// ---------------- fused flash attention ----------------
// grid (F/128, HEADS, B); 8 waves/block; wave = 16 query rows, streams J in chunks of 32.
// Per chunk: 4 WMMA (S), DPP max-reduce, exp, LDS transpose bounce,
//            1 WMMA (row-sum of P), 4 WMMA (O += P@V).
__global__ __launch_bounds__(256) void attn_kernel(const _Float16* __restrict__ qh,  // [B*F, INNER], pre-scaled
                                                   const _Float16* __restrict__ kh,  // [B*J, INNER]
                                                   const _Float16* __restrict__ vT,  // [B, INNER, J]
                                                   _Float16* __restrict__ oh) {      // [B*F, INNER]
    __shared__ __align__(16) _Float16 plds[8][16 * 32];   // 1KB per wave P-tile bounce
    const int w    = threadIdx.x >> 5;
    const int lane = threadIdx.x & 31;
    const int n = lane & 15, half = lane >> 4;
    const int h = blockIdx.y, b = blockIdx.z;
    const int f0 = blockIdx.x * 128 + w * 16;

    const _Float16* qbase = qh + ((size_t)b * F_ + f0) * INNER + h * DHEAD;
    const v16h qa0 = load_fragA(qbase, INNER, 0, 0);
    const v16h qa1 = load_fragA(qbase, INNER, 0, 32);
    const v16h ones = ones_frag();

    v8f o0 = {}, o1 = {}, o2 = {}, o3 = {};
    float mrun[8], lrun[8];
#pragma unroll
    for (int r = 0; r < 8; ++r) { mrun[r] = -1e30f; lrun[r] = 0.f; }

    const _Float16* vbase = vT + ((size_t)b * INNER + h * DHEAD) * J_;
    _Float16* pl = &plds[w][0];

    for (int j0 = 0; j0 < J_; j0 += 32) {
        // ---- S = (q*scale) @ k^T : 16 x 32 tile, K = DHEAD = 64 ----
        const _Float16* kb = kh + ((size_t)b * J_ + j0) * INNER + h * DHEAD;
        v8f s0 = {}, s1 = {};
        s0 = wmma16(qa0, load_fragB(kb, INNER, 0, 0),   s0);
        s0 = wmma16(qa1, load_fragB(kb, INNER, 0, 32),  s0);
        s1 = wmma16(qa0, load_fragB(kb, INNER, 16, 0),  s1);
        s1 = wmma16(qa1, load_fragB(kb, INNER, 16, 32), s1);

        // ---- online softmax: DPP max-reduce, rescale, exponentiate ----
#pragma unroll
        for (int r = 0; r < 8; ++r) {
            float mx = redmax16(fmaxf(s0[r], s1[r]));
            float mnew = fmaxf(mrun[r], mx);
            float corr = __expf(mrun[r] - mnew);
            mrun[r] = mnew;
            lrun[r] *= corr;
            o0[r] *= corr; o1[r] *= corr; o2[r] *= corr; o3[r] *= corr;
            float e0 = __expf(s0[r] - mnew);
            float e1 = __expf(s1[r] - mnew);
            int row = r + 8 * half;
            pl[row * 32 + n]      = (_Float16)e0;   // C-layout -> row-major 16x32
            pl[row * 32 + 16 + n] = (_Float16)e1;
        }
        __builtin_amdgcn_wave_barrier();
        asm volatile("s_wait_dscnt 0" ::: "memory");   // same-wave LDS RAW fence
        v16h pa = load_fragA(pl, 32, 0, 0);
        __builtin_amdgcn_wave_barrier();

        // ---- row sums of P via WMMA against all-ones (replaces shuffle reduce) ----
        v8f rs = {};
        rs = wmma16(pa, ones, rs);
#pragma unroll
        for (int r = 0; r < 8; ++r) lrun[r] += rs[r];

        // ---- O += P @ V : 4 d-tiles of 16, K = 32 (j chunk) ----
        o0 = wmma16(pa, load_fragB(vbase, J_, 0,  j0), o0);
        o1 = wmma16(pa, load_fragB(vbase, J_, 16, j0), o1);
        o2 = wmma16(pa, load_fragB(vbase, J_, 32, j0), o2);
        o3 = wmma16(pa, load_fragB(vbase, J_, 48, j0), o3);
    }

    // ---- epilogue: O / l -> f16 ----
    _Float16* ob = oh + ((size_t)b * F_ + f0) * INNER + h * DHEAD;
#pragma unroll
    for (int r = 0; r < 8; ++r) {
        float invl = 1.0f / lrun[r];
        int row = r + 8 * half;
        ob[(size_t)row * INNER + n]      = (_Float16)(o0[r] * invl);
        ob[(size_t)row * INNER + 16 + n] = (_Float16)(o1[r] * invl);
        ob[(size_t)row * INNER + 32 + n] = (_Float16)(o2[r] * invl);
        ob[(size_t)row * INNER + 48 + n] = (_Float16)(o3[r] * invl);
    }
}

// ---------------- host orchestration ----------------
extern "C" void kernel_launch(void* const* d_in, const int* in_sizes, int n_in,
                              void* d_out, int out_size, void* d_ws, size_t ws_size,
                              hipStream_t stream) {
    (void)in_sizes; (void)n_in; (void)out_size; (void)ws_size;
    const float* x     = (const float*)d_in[0];
    const float* tab_x = (const float*)d_in[1];
    const float* tab_g = (const float*)d_in[2];
    const float* tab_b = (const float*)d_in[3];
    const float* vid_g = (const float*)d_in[4];
    const float* vid_b = (const float*)d_in[5];
    const float* Wq    = (const float*)d_in[6];
    const float* Wk    = (const float*)d_in[7];
    const float* Wv    = (const float*)d_in[8];
    const float* q_g   = (const float*)d_in[9];
    const float* q_b   = (const float*)d_in[10];
    const float* k_g   = (const float*)d_in[11];
    const float* k_b   = (const float*)d_in[12];
    const float* Wo    = (const float*)d_in[13];
    const float* bo    = (const float*)d_in[14];

    // workspace layout (256B aligned)
    char* ws = (char*)d_ws;
    size_t off = 0;
    auto alloc = [&](size_t bytes) { char* p = ws + off; off = (off + bytes + 255) & ~(size_t)255; return p; };
    _Float16* WqT = (_Float16*)alloc((size_t)INNER * DIM * 2);
    _Float16* WkT = (_Float16*)alloc((size_t)INNER * CTX * 2);
    _Float16* WvT = (_Float16*)alloc((size_t)INNER * CTX * 2);
    _Float16* WoT = (_Float16*)alloc((size_t)DIM * INNER * 2);
    _Float16* xln = (_Float16*)alloc((size_t)B_ * F_ * DIM * 2);
    _Float16* kv  = (_Float16*)alloc((size_t)B_ * J_ * CTX * 2);
    _Float16* qh  = (_Float16*)alloc((size_t)B_ * F_ * INNER * 2);
    _Float16* khf = (_Float16*)alloc((size_t)B_ * J_ * INNER * 2);
    _Float16* vTr = (_Float16*)alloc((size_t)B_ * INNER * J_ * 2);
    _Float16* ah  = (_Float16*)alloc((size_t)B_ * F_ * INNER * 2);
    float*    pre = (float*)   alloc((size_t)B_ * F_ * INNER * 4);   // reused q/k/v scratch

    const float qscale = 0.125f;  // DHEAD^-0.5

    // 1. weights -> transposed f16
    wT_kernel<<<(INNER * DIM + 255) / 256, 256, 0, stream>>>(Wq, WqT, DIM, INNER);
    wT_kernel<<<(INNER * CTX + 255) / 256, 256, 0, stream>>>(Wk, WkT, CTX, INNER);
    wT_kernel<<<(INNER * CTX + 255) / 256, 256, 0, stream>>>(Wv, WvT, CTX, INNER);
    wT_kernel<<<(DIM * INNER + 255) / 256, 256, 0, stream>>>(Wo, WoT, INNER, DIM);

    // 2. input layernorms -> f16
    ln_kernel<<<B_ * J_, 256, 0, stream>>>(tab_x, kv, tab_g, tab_b, CTX, 1.0f);
    ln_kernel<<<B_ * F_, 256, 0, stream>>>(x, xln, vid_g, vid_b, DIM, 1.0f);

    // 3. q = LN(xln @ Wq) * scale
    gemm_kernel<false><<<dim3(INNER / 256, (B_ * F_) / 64), 256, 0, stream>>>(
        xln, WqT, pre, nullptr, B_ * F_, INNER, DIM);
    ln_kernel<<<B_ * F_, 256, 0, stream>>>(pre, qh, q_g, q_b, INNER, qscale);

    // 4. k = LN(kv @ Wk)
    gemm_kernel<false><<<dim3(INNER / 256, (B_ * J_) / 64), 256, 0, stream>>>(
        kv, WkT, pre, nullptr, B_ * J_, INNER, CTX);
    ln_kernel<<<B_ * J_, 256, 0, stream>>>(pre, khf, k_g, k_b, INNER, 1.0f);

    // 5. v = kv @ Wv, transposed f16 for B-fragment loads
    gemm_kernel<false><<<dim3(INNER / 256, (B_ * J_) / 64), 256, 0, stream>>>(
        kv, WvT, pre, nullptr, B_ * J_, INNER, CTX);
    vT_kernel<<<(unsigned)(((size_t)B_ * J_ * INNER + 255) / 256), 256, 0, stream>>>(pre, vTr);

    // 6. fused flash attention
    attn_kernel<<<dim3(F_ / 128, HEADS, B_), 256, 0, stream>>>(qh, khf, vTr, ah);

    // 7. out = attn_out @ Wo + bo  (fp32 to d_out)
    gemm_kernel<true><<<dim3(DIM / 256, (B_ * F_) / 64), 256, 0, stream>>>(
        ah, WoT, (float*)d_out, bo, B_ * F_, DIM, INNER);
}
